// DotProductAttention_15504831939087
// MI455X (gfx1250) — matrix operand order repaired and checked by reading, not compiled, not run on hardware
//
#include <hip/hip_runtime.h>
#include <hip/hip_bf16.h>
#include <math.h>

// Problem constants (match reference)
#define BATCH 16
#define LQ    2048
#define LK    2048
#define DIM   64

// Tiling
#define BQ    128      // query rows per workgroup (8 waves x 16 rows)
#define BK    64       // key tile
#define NWAVE 8
#define NTHREADS 256

typedef _Float16 v16h __attribute__((ext_vector_type(16)));
typedef _Float16 v8h  __attribute__((ext_vector_type(8)));
typedef float    v8f  __attribute__((ext_vector_type(8)));

// Load a WMMA A-operand (16-bit 16x32, MxK) fragment for this lane.
// rowp points at this lane's row (row = ln) of a row-major f16 matrix.
// A layout: lanes 0-15 hold K = ks..ks+15, lanes 16-31 hold K = ks+16..ks+31 (contiguous, matches ld_b).
__device__ __forceinline__ v16h ld_a(const _Float16* rowp, int ks, int half) {
    v8h lo = *(const v8h*)(rowp + ks + half * 16);
    v8h hi = *(const v8h*)(rowp + ks + half * 16 + 8);
    return __builtin_shufflevector(lo, hi, 0,1,2,3,4,5,6,7,8,9,10,11,12,13,14,15);
}

// Load a WMMA B-operand (16-bit 32x16, KxN) fragment for this lane.
// rowp points at this lane's "column source row" (row-major storage where
// rows correspond to N and columns to K). B layout: lanes 0-15 hold K=0..15,
// lanes 16-31 hold K=16..31 -> contiguous 16 f16 starting at koff.
__device__ __forceinline__ v16h ld_b(const _Float16* rowp, int koff) {
    v8h lo = *(const v8h*)(rowp + koff);
    v8h hi = *(const v8h*)(rowp + koff + 8);
    return __builtin_shufflevector(lo, hi, 0,1,2,3,4,5,6,7,8,9,10,11,12,13,14,15);
}

__device__ __forceinline__ v8f wmma_f16(v16h a, v16h b, v8f c) {
    return __builtin_amdgcn_wmma_f32_16x16x32_f16(false, a, false, b, (short)0, c,
                                                  false, false);
}

__global__ __launch_bounds__(NTHREADS)
void flash_attn_f16wmma(const float* __restrict__ q,
                        const float* __restrict__ k,
                        const float* __restrict__ v,
                        const int*   __restrict__ valid_num,
                        float*       __restrict__ out) {
    __shared__ __attribute__((aligned(16))) _Float16 sQ[BQ * DIM];      // 16 KB
    __shared__ __attribute__((aligned(16))) _Float16 sK[BK * DIM];      // 8 KB
    __shared__ __attribute__((aligned(16))) _Float16 sVT[DIM * BK];     // 8 KB (V transposed)
    __shared__ __attribute__((aligned(16))) _Float16 sP[NWAVE * 16 * BK]; // 16 KB

    const int tid  = threadIdx.x;
    const int wave = tid >> 5;
    const int lane = tid & 31;
    const int half = lane >> 4;
    const int ln   = lane & 15;

    const int b     = blockIdx.y;
    const int qBase = blockIdx.x * BQ;
    const int vnum  = valid_num[b];
    const int ktiles = (vnum + BK - 1) / BK;   // skip fully-masked tail tiles

    const float scale = 0.125f;                 // 1/sqrt(64)

    // ---- Stage Q (fp32 -> f16) into LDS, then pull A-operands into registers.
    const float* qg = q + ((size_t)b * LQ + qBase) * DIM;
    #pragma unroll
    for (int i = 0; i < (BQ * DIM) / NTHREADS; ++i) {
        int idx = tid + i * NTHREADS;
        sQ[idx] = (_Float16)qg[idx];
    }
    __syncthreads();

    const _Float16* qrow = sQ + (wave * 16 + ln) * DIM;
    v16h aq0 = ld_a(qrow, 0, half);
    v16h aq1 = ld_a(qrow, 32, half);

    // ---- Persistent per-wave state
    v8f O[4];
    #pragma unroll
    for (int a4 = 0; a4 < 4; ++a4) O[a4] = (v8f){0,0,0,0,0,0,0,0};
    float mrow[8], lrow[8];
    #pragma unroll
    for (int r = 0; r < 8; ++r) { mrow[r] = -INFINITY; lrow[r] = 0.0f; }

    _Float16* pw = sP + wave * 16 * BK;   // this wave's P scratch (16 x 64 f16)

    for (int kt = 0; kt < ktiles; ++kt) {
        __syncthreads();   // previous compute done before overwriting sK/sVT

        const float* kg = k + ((size_t)b * LK + kt * BK) * DIM;
        const float* vg = v + ((size_t)b * LK + kt * BK) * DIM;
        #pragma unroll
        for (int i = 0; i < (BK * DIM) / NTHREADS; ++i) {
            int idx = tid + i * NTHREADS;
            int r = idx >> 6, c = idx & 63;
            sK[idx]          = (_Float16)kg[idx];
            sVT[c * BK + r]  = (_Float16)vg[idx];   // transpose V at staging
        }
        __syncthreads();

        // Prefetch next K/V tile while we compute (global_prefetch_b8).
        if (kt + 1 < ktiles) {
            __builtin_prefetch(kg + BK * DIM + tid * 16, 0, 0);
            __builtin_prefetch(vg + BK * DIM + tid * 16, 0, 0);
        }

        // ---- S = Q * K^T  (16 x 64 tile = 4 accumulators)
        v8f S[4];
        #pragma unroll
        for (int a4 = 0; a4 < 4; ++a4) S[a4] = (v8f){0,0,0,0,0,0,0,0};
        #pragma unroll
        for (int ks = 0; ks < DIM; ks += 32) {
            v16h a = (ks == 0) ? aq0 : aq1;
            #pragma unroll
            for (int a4 = 0; a4 < 4; ++a4) {
                v16h bk = ld_b(sK + (a4 * 16 + ln) * DIM, ks + half * 16);
                S[a4] = wmma_f16(a, bk, S[a4]);
            }
        }

        // ---- Mask + scale (branch-free: add -inf where col >= valid)
        const int colBase = kt * BK;
        #pragma unroll
        for (int a4 = 0; a4 < 4; ++a4) {
            float sel = ((colBase + a4 * 16 + ln) < vnum) ? 0.0f : -INFINITY;
            #pragma unroll
            for (int r = 0; r < 8; ++r) S[a4][r] = S[a4][r] * scale + sel;
        }

        // ---- Online softmax per row (row = r + half*8; cols striped over 16 lanes)
        #pragma unroll
        for (int r = 0; r < 8; ++r) {
            float mx = fmaxf(fmaxf(S[0][r], S[1][r]), fmaxf(S[2][r], S[3][r]));
            mx = fmaxf(mx, __shfl_xor(mx, 1));
            mx = fmaxf(mx, __shfl_xor(mx, 2));
            mx = fmaxf(mx, __shfl_xor(mx, 4));
            mx = fmaxf(mx, __shfl_xor(mx, 8));
            float mnew  = fmaxf(mrow[r], mx);
            float alpha = __expf(mrow[r] - mnew);
            mrow[r] = mnew;

            float rs = 0.0f;
            #pragma unroll
            for (int a4 = 0; a4 < 4; ++a4) {
                float p = __expf(S[a4][r] - mnew);
                S[a4][r] = p;
                rs += p;
            }
            rs += __shfl_xor(rs, 1);
            rs += __shfl_xor(rs, 2);
            rs += __shfl_xor(rs, 4);
            rs += __shfl_xor(rs, 8);
            lrow[r] = lrow[r] * alpha + rs;

            #pragma unroll
            for (int a4 = 0; a4 < 4; ++a4) O[a4][r] *= alpha;
        }

        // ---- Transpose P from C-layout to A-layout via wave-private LDS
        #pragma unroll
        for (int a4 = 0; a4 < 4; ++a4) {
            #pragma unroll
            for (int r = 0; r < 8; ++r) {
                pw[(r + half * 8) * BK + a4 * 16 + ln] = (_Float16)S[a4][r];
            }
        }
        // LDS ops from a wave are in-order; wait makes it explicit and fences compiler.
        asm volatile("s_wait_dscnt 0x0" ::: "memory");

        // ---- O += P * V   (K dim = 64 keys, two 32-key steps)
        #pragma unroll
        for (int kk = 0; kk < BK; kk += 32) {
            v16h ap = ld_a(pw + ln * BK, kk, half);
            #pragma unroll
            for (int a4 = 0; a4 < 4; ++a4) {
                v16h bv = ld_b(sVT + (a4 * 16 + ln) * BK, kk + half * 16);
                O[a4] = wmma_f16(ap, bv, O[a4]);
            }
        }
    }

    // ---- Epilogue: O / l, write fp32 output
    float* og = out + ((size_t)b * LQ + qBase + wave * 16) * DIM;
    #pragma unroll
    for (int a4 = 0; a4 < 4; ++a4) {
        #pragma unroll
        for (int r = 0; r < 8; ++r) {
            int m = r + half * 8;
            og[m * DIM + a4 * 16 + ln] = O[a4][r] / lrow[r];
        }
    }
}

extern "C" void kernel_launch(void* const* d_in, const int* in_sizes, int n_in,
                              void* d_out, int out_size, void* d_ws, size_t ws_size,
                              hipStream_t stream) {
    const float* q = (const float*)d_in[0];
    const float* k = (const float*)d_in[1];
    const float* v = (const float*)d_in[2];
    const int* valid = (const int*)d_in[3];
    float* out = (float*)d_out;

    dim3 grid(LQ / BQ, BATCH, 1);
    dim3 block(NTHREADS, 1, 1);
    hipLaunchKernelGGL(flash_attn_f16wmma, grid, block, 0, stream,
                       q, k, v, valid, out);
}
